// VectorQuantizer_25701084299871
// MI455X (gfx1250) — matrix operand order repaired and checked by reading, not compiled, not run on hardware
//
#include <hip/hip_runtime.h>
#include <hip/hip_bf16.h>

// Problem constants (from reference): x[32,64,64,64] NCHW fp32, W[512,64] fp32.
#define BATCH 32
#define CH    64      // == D
#define HH    64
#define WW    64
#define KCODE 512
#define DDIM  64

#define NTOK   (BATCH * HH * WW)        // 131072 tokens
#define NSTRIP (NTOK / 16)              // 8192 strips of 16 tokens along w
#define HW     (HH * WW)                // 4096 (feature stride in floats)

// LDS codebook layout: per code, 32 packed bf16x2 pairs (64 features),
// padded to 36 u32 (144B) per code -> 16 consecutive codes hit 16 distinct
// banks for ds_load_b128 (c*36 mod 64 distinct for c=0..15).
#define CSTRIDE_U32 36
#define HI_BANK_U32 (KCODE * CSTRIDE_U32)   // 18432 u32 = 72KB per bank

typedef __attribute__((ext_vector_type(16))) __bf16 v16bf;
typedef __attribute__((ext_vector_type(8)))  float  v8f;

union BFrag { v16bf v; uint4 q[2]; };
union PackBF { uint32_t u; __bf16 h[2]; };

static __device__ __forceinline__ void split_bf16(float a, __bf16& hi, __bf16& lo) {
    hi = (__bf16)a;
    lo = (__bf16)(a - (float)hi);
}

// ---------------------------------------------------------------------------
// Kernel 0: exact fp32 codebook norms ||w_k||^2  (512 values)
// ---------------------------------------------------------------------------
__global__ void vq_wnorm(const float* __restrict__ W, float* __restrict__ wn) {
    int k = blockIdx.x * 256 + threadIdx.x;
    if (k < KCODE) {
        const float4* row = (const float4*)(W + (size_t)k * DDIM);
        float s = 0.f;
#pragma unroll
        for (int i = 0; i < DDIM / 4; ++i) {
            float4 v = row[i];
            s += v.x * v.x + v.y * v.y + v.z * v.z + v.w * v.w;
        }
        wn[k] = s;
    }
}

// ---------------------------------------------------------------------------
// Kernel 1: distances via bf16 hi/lo-split WMMA + per-token argmin.
// Block = 256 threads = 8 waves; each wave owns one strip of 16 tokens
// (fixed b,h; w0..w0+15 -> contiguous global loads per feature).
// Whole codebook staged once per block in LDS as packed hi/lo bf16
// (B-fragment layout), ~146KB dynamic LDS -> 2 blocks per 320KB WGP pool.
// dist' = ||w||^2 - 2*(xh*wh + xh*wl + xl*wh)   (||x||^2 const per token)
// ---------------------------------------------------------------------------
__global__ void __launch_bounds__(256)
vq_argmin2(const float* __restrict__ x, const float* __restrict__ W,
           const float* __restrict__ wnorm_g, int* __restrict__ idx_out) {
    extern __shared__ uint32_t smem[];
    uint32_t* hiB = smem;                       // packed bf16 hi pairs
    uint32_t* loB = smem + HI_BANK_U32;         // packed bf16 lo pairs
    float*    wnL = (float*)(smem + 2 * HI_BANK_U32);

    const int tid  = threadIdx.x;
    const int lane = tid & 31;
    const int wave = tid >> 5;
    const int hl   = lane >> 4;                 // half of wave (0/1)
    const int lm   = lane & 15;

    // ---- Stage codebook (hi/lo split) + norms into LDS ------------------
    for (int i = tid; i < KCODE * 32; i += 256) {
        const int c = i >> 5, p = i & 31;       // pair p = features 2p,2p+1
        const float2 wv = *(const float2*)(W + (size_t)c * DDIM + 2 * p);
        PackBF ph, pl;
        split_bf16(wv.x, ph.h[0], pl.h[0]);
        split_bf16(wv.y, ph.h[1], pl.h[1]);
        hiB[c * CSTRIDE_U32 + p] = ph.u;
        loB[c * CSTRIDE_U32 + p] = pl.u;
    }
    for (int i = tid; i < KCODE; i += 256) wnL[i] = wnorm_g[i];
    __syncthreads();

    // ---- Build A fragments (16 tokens x 64 features, hi/lo) -------------
    // A 16x32 layout (contiguous, matches B): lane half hl, VGPR v -> K = 16*hl + 2v;
    // chunk q covers features 32q..32q+31.
    const int s  = blockIdx.x * 8 + wave;       // strip id, < NSTRIP
    const int w0 = (s & 3) * 16;
    const int h  = (s >> 2) & (HH - 1);
    const int b  = s >> 8;
    const int t  = lm;                          // token (= M row) of this lane
    const float* xb = x + (((size_t)b * CH) * HH + h) * WW + w0 + t;

    v16bf ah[2], al[2];
#pragma unroll
    for (int q = 0; q < 2; ++q)
#pragma unroll
        for (int g = 0; g < 2; ++g)
#pragma unroll
            for (int j = 0; j < 4; ++j) {
                const int f = 32 * q + 16 * hl + 8 * g + 2 * j; // feature (K)
                const int e = 8 * g + 2 * j;                    // vector elem
                const float a0 = xb[(size_t)f * HW];
                const float a1 = xb[(size_t)(f + 1) * HW];
                __bf16 h0, l0, h1, l1;
                split_bf16(a0, h0, l0);
                split_bf16(a1, h1, l1);
                ah[q][e] = h0; ah[q][e + 1] = h1;
                al[q][e] = l0; al[q][e + 1] = l1;
            }

    // ---- Sweep 32 code tiles; running per-lane (min, argmin) ------------
    float minv[8];
    int   mini[8];
#pragma unroll
    for (int r = 0; r < 8; ++r) { minv[r] = 3.4e38f; mini[r] = 0; }

    for (int ct = 0; ct < KCODE / 16; ++ct) {
        const int c = ct * 16 + lm;             // this lane's code (N col)
        const float wn = wnL[c];

        // Dense-B layout: lane = N column, 16 consecutive K per lane
        // -> 8 consecutive u32 pairs starting at pair 16q + 8*hl.
        BFrag bh[2], bl[2];
#pragma unroll
        for (int q = 0; q < 2; ++q) {
            const int off = c * CSTRIDE_U32 + q * 16 + hl * 8;
            bh[q].q[0] = *(const uint4*)&hiB[off];
            bh[q].q[1] = *(const uint4*)&hiB[off + 4];
            bl[q].q[0] = *(const uint4*)&loB[off];
            bl[q].q[1] = *(const uint4*)&loB[off + 4];
        }

        v8f accHH = {};  // xh*wh
        v8f accX  = {};  // xh*wl + xl*wh  (lo*lo dropped)
#pragma unroll
        for (int q = 0; q < 2; ++q) {
            accHH = __builtin_amdgcn_wmma_f32_16x16x32_bf16(
                false, ah[q], false, bh[q].v, (short)0, accHH, false, false);
            accX  = __builtin_amdgcn_wmma_f32_16x16x32_bf16(
                false, ah[q], false, bl[q].v, (short)0, accX,  false, false);
            accX  = __builtin_amdgcn_wmma_f32_16x16x32_bf16(
                false, al[q], false, bh[q].v, (short)0, accX,  false, false);
        }

#pragma unroll
        for (int r = 0; r < 8; ++r) {
            const float dot = accHH[r] + accX[r];
            const float d   = wn - 2.0f * dot;
            // strict '<' + ascending c keeps the lowest index on ties
            const bool take = d < minv[r];
            minv[r] = take ? d : minv[r];
            mini[r] = take ? c : mini[r];
        }
    }

    // ---- Branchless reduce across the 16 code-lanes of each half --------
    // xor masks 1,2,4,8 keep the two 16-lane halves separate (wave32).
#pragma unroll
    for (int m = 1; m <= 8; m <<= 1)
#pragma unroll
        for (int r = 0; r < 8; ++r) {
            const float ov = __shfl_xor(minv[r], m, 32);
            const int   oi = __shfl_xor(mini[r], m, 32);
            // bitwise ops (no short-circuit) -> v_cmp/v_cndmask, no branches
            const bool take = (ov < minv[r]) |
                              ((ov == minv[r]) & (oi < mini[r]));
            minv[r] = take ? ov : minv[r];
            mini[r] = take ? oi : mini[r];
        }

    // token M = r + 8*hl ; flat token id n = s*16 + M
#pragma unroll
    for (int r = 0; r < 8; ++r)
        if (lm == r) idx_out[s * 16 + r + 8 * hl] = mini[r];
}

// ---------------------------------------------------------------------------
// Kernel 2: gather exact fp32 rows, scatter to NCHW (coalesced writes along w,
// codebook reads random-row but L2-resident at 128KB).
// ---------------------------------------------------------------------------
__global__ void vq_gather(const float* __restrict__ W,
                          const int* __restrict__ idxb,
                          float* __restrict__ out) {
    const size_t o = (size_t)blockIdx.x * 256 + threadIdx.x; // NCHW linear
    const int w = (int)(o & 63);
    const int h = (int)((o >> 6) & 63);
    const int d = (int)((o >> 12) & 63);
    const int b = (int)(o >> 18);
    const int n = (b * HH + h) * WW + w;
    out[o] = W[(size_t)idxb[n] * DDIM + d];
}

// ---------------------------------------------------------------------------
extern "C" void kernel_launch(void* const* d_in, const int* in_sizes, int n_in,
                              void* d_out, int out_size, void* d_ws, size_t ws_size,
                              hipStream_t stream) {
    const float* x = (const float*)d_in[0];   // [32,64,64,64] fp32
    const float* W = (const float*)d_in[1];   // [512,64] fp32
    float* out = (float*)d_out;               // [32,64,64,64] fp32

    // workspace: [0,2KB) code norms, then N ints of argmin indices (512KB)
    float* wnorm = (float*)d_ws;
    int*   idxb  = (int*)((char*)d_ws + KCODE * sizeof(float));

    vq_wnorm<<<(KCODE + 255) / 256, 256, 0, stream>>>(W, wnorm);

    const size_t ldsBytes = (size_t)(2 * HI_BANK_U32) * 4 + KCODE * 4; // ~146KB
    vq_argmin2<<<NSTRIP / 8, 256, ldsBytes, stream>>>(x, W, wnorm, idxb);

    const size_t nOut = (size_t)NTOK * DDIM;  // 8388608
    vq_gather<<<(unsigned)((nOut + 255) / 256), 256, 0, stream>>>(W, idxb, out);
}
